// AttentionNet_35777077576575
// MI455X (gfx1250) — compile-verified, matrix operand order checked by reading, not run on hardware
//
#include <hip/hip_runtime.h>

#define B_DIM   2048
#define N_OBJ   128
#define D_OBJ   64
#define D_MODEL 512
#define LDS_LD  520   // padded row stride (halfs): 1040B % 256B = 16 -> conflict-free rows

typedef __attribute__((ext_vector_type(16))) _Float16 v16h;
typedef __attribute__((ext_vector_type(8)))  _Float16 v8h;
typedef __attribute__((ext_vector_type(8)))  float    v8f;
typedef __attribute__((ext_vector_type(4)))  int      v4i;

#if __has_builtin(__builtin_amdgcn_global_load_async_to_lds_b128)
typedef __attribute__((address_space(1))) v4i* as1_v4i_ptr;
typedef __attribute__((address_space(3))) v4i* as3_v4i_ptr;
#endif

// K index covered by element e of a 16-bit A/B WMMA fragment, for lane-half h.
// Per ISA 7.12.2: VGPR0..3 hold K = h*8 + 0..7, VGPR4..7 hold K = 16 + h*8 + 0..7.
__device__ __forceinline__ int a_kmap(int e, int h) {
  return (e < 8) ? (h * 8 + e) : (16 + h * 8 + (e - 8));
}

// ---------------------------------------------------------------------------
// Pre-pack a row-major f32 [K x N] weight matrix into per-lane B-fragment
// layout: pk[((ntile*kSteps + ks)*32 + lane)*16 + e]. Makes GEMM B loads two
// contiguous b128 reads per lane.
// ---------------------------------------------------------------------------
__global__ void pack_b_kernel(const float* __restrict__ src, _Float16* __restrict__ dst,
                              int K, int N) {
  int tid = blockIdx.x * blockDim.x + threadIdx.x;
  int kSteps = K >> 5;
  int total = (N >> 4) * kSteps * 32;
  if (tid >= total) return;
  int lane  = tid & 31;
  int ks    = (tid >> 5) % kSteps;
  int ntile = tid / (32 * kSteps);
  int hlf   = lane >> 4;
  int n     = (ntile << 4) + (lane & 15);
  _Float16* out = dst + ((size_t)(ntile * kSteps + ks) * 32 + lane) * 16;
#pragma unroll
  for (int e = 0; e < 16; ++e) {
    int k = (ks << 5) + a_kmap(e, hlf);
    out[e] = (_Float16)src[(size_t)k * N + n];
  }
}

// ---- fragment loaders ------------------------------------------------------
__device__ __forceinline__ v16h load_a_f32(const float* __restrict__ A, int row0, int ld, int k0) {
  int lane = threadIdx.x & 31;
  const float* p = A + (size_t)(row0 + (lane & 15)) * ld + k0 + ((lane >> 4) << 3);
  v16h a;
#pragma unroll
  for (int e = 0; e < 8; ++e) { a[e] = (_Float16)p[e]; a[e + 8] = (_Float16)p[e + 16]; }
  return a;
}

__device__ __forceinline__ v16h load_a_lds(const _Float16* sA, int k0) {
  int lane = threadIdx.x & 31;
  const _Float16* p = sA + (size_t)(lane & 15) * LDS_LD + k0 + ((lane >> 4) << 3);
  union { v16h v; v8h h[2]; } u;
  u.h[0] = *(const v8h*)p;        // 16B aligned: row*1040B + k0*2 + 16B*half
  u.h[1] = *(const v8h*)(p + 16);
  return u.v;
}

__device__ __forceinline__ v16h load_b_pk(const _Float16* __restrict__ pk, int tileIdx,
                                          int kSteps, int ks) {
  int lane = threadIdx.x & 31;
  const v8h* p = (const v8h*)(pk + ((size_t)(tileIdx * kSteps + ks) * 32 + lane) * 16);
  union { v16h v; v8h h[2]; } u;
  u.h[0] = p[0]; u.h[1] = p[1];
  return u.v;
}

// ---------------------------------------------------------------------------
// Kernel 1: emb = x @ conv_w + b, then mask -> x_real (f16).
// Grid: (B*N/16, 512/128), block 256 (8 waves; wave w owns cols w*16..w*16+15).
// ---------------------------------------------------------------------------
__global__ void emb_mask_kernel(const float* __restrict__ x, const float* __restrict__ mask,
                                const float* __restrict__ bias, const _Float16* __restrict__ pkW,
                                _Float16* __restrict__ xreal) {
  int wave = threadIdx.x >> 5, lane = threadIdx.x & 31, hlf = lane >> 4;
  int row0 = blockIdx.x << 4;
  int col  = (blockIdx.y << 7) + (wave << 4) + (lane & 15);
  int tileIdx = (blockIdx.y << 3) + wave;
  v8f c = {};
#pragma unroll
  for (int ks = 0; ks < 2; ++ks) {
    v16h a = load_a_f32(x, row0, D_OBJ, ks << 5);
    v16h b = load_b_pk(pkW, tileIdx, 2, ks);
    c = __builtin_amdgcn_wmma_f32_16x16x32_f16(false, a, false, b, (short)0, c, false, false);
  }
  float bv = bias[col];
#pragma unroll
  for (int r = 0; r < 8; ++r) {
    int row = row0 + r + (hlf << 3);             // C/D layout: M = r + 8*half
    float v = (c[r] + bv) * mask[row];           // mask flat index == global row
    xreal[(size_t)row * D_MODEL + col] = (_Float16)v;
  }
}

// ---------------------------------------------------------------------------
// Kernel 2: query[b,:] = sum_n x_real[b,n,:] / (sum_n mask + 1e-5)
// ---------------------------------------------------------------------------
__global__ void query_kernel(const _Float16* __restrict__ xreal, const float* __restrict__ mask,
                             float* __restrict__ query) {
  int b = blockIdx.x;
  float denom = 1e-5f;
  for (int n = 0; n < N_OBJ; ++n) denom += mask[b * N_OBJ + n];
  float inv = 1.0f / denom;
  for (int col = threadIdx.x; col < D_MODEL; col += blockDim.x) {
    float s = 0.f;
    const _Float16* p = xreal + (size_t)b * N_OBJ * D_MODEL + col;
    for (int n = 0; n < N_OBJ; ++n) s += (float)p[(size_t)n * D_MODEL];
    query[(size_t)b * D_MODEL + col] = s * inv;
  }
}

// ---------------------------------------------------------------------------
// Kernel 3: q_proj = query @ Uq   (2048x512 @ 512x512)
// ---------------------------------------------------------------------------
__global__ void qproj_kernel(const float* __restrict__ query, const _Float16* __restrict__ pkUq,
                             float* __restrict__ qproj) {
  int wave = threadIdx.x >> 5, lane = threadIdx.x & 31, hlf = lane >> 4;
  int row0 = blockIdx.x << 4;
  int col  = (blockIdx.y << 7) + (wave << 4) + (lane & 15);
  int tileIdx = (blockIdx.y << 3) + wave;
  v8f c = {};
#pragma unroll
  for (int ks = 0; ks < 16; ++ks) {
    v16h a = load_a_f32(query, row0, D_MODEL, ks << 5);
    v16h b = load_b_pk(pkUq, tileIdx, 16, ks);
    c = __builtin_amdgcn_wmma_f32_16x16x32_f16(false, a, false, b, (short)0, c, false, false);
  }
#pragma unroll
  for (int r = 0; r < 8; ++r) {
    int row = row0 + r + (hlf << 3);
    qproj[(size_t)row * D_MODEL + col] = c[r];
  }
}

// ---------------------------------------------------------------------------
// Kernel 4: logits[b,n] = Ua . tanh(q_proj[b] + x_real[b,n] @ Ur)
// One block = 16 rows (one b, 16 n). The 16x512 f16 A tile is staged into LDS
// once per block via the CDNA5 async global->LDS path (ASYNCcnt), then all
// 8 waves build A fragments with ds_load_b128 (row stride padded to kill bank
// conflicts). 8 waves x 4 col-tiles cover all 512 cols; the 16x512 r_proj
// tile lives only in accumulator registers.
// ---------------------------------------------------------------------------
__global__ void logits_kernel(const _Float16* __restrict__ xreal, const _Float16* __restrict__ pkUr,
                              const float* __restrict__ qproj, const float* __restrict__ Ua,
                              float* __restrict__ logits) {
  __shared__ _Float16 sA[16 * LDS_LD];   // ~16.6 KB, padded rows
  __shared__ float slog[16];
  int tid = threadIdx.x;
  int wave = tid >> 5, lane = tid & 31, hlf = lane >> 4;
  int row0 = blockIdx.x << 4;          // global row = b*128 + n
  int bIdx = row0 >> 7;
  if (tid < 16) slog[tid] = 0.f;

  // Stage A tile: 16 rows x 512 halfs = 1024 chunks of 16B; 4 chunks/thread.
  {
    const _Float16* gsrc = xreal + (size_t)row0 * D_MODEL;
#pragma unroll
    for (int i = 0; i < 4; ++i) {
      int chunk = tid + (i << 8);                  // 0..1023
      int r = chunk >> 6;                          // row 0..15
      int o = (chunk & 63) << 3;                   // half-offset in row
#if __has_builtin(__builtin_amdgcn_global_load_async_to_lds_b128)
      __builtin_amdgcn_global_load_async_to_lds_b128(
          (as1_v4i_ptr)(gsrc + (size_t)r * D_MODEL + o),
          (as3_v4i_ptr)(sA + (size_t)r * LDS_LD + o),
          0, 0);
#else
      *(v8h*)(sA + (size_t)r * LDS_LD + o) = *(const v8h*)(gsrc + (size_t)r * D_MODEL + o);
#endif
    }
#if __has_builtin(__builtin_amdgcn_global_load_async_to_lds_b128)
  #if __has_builtin(__builtin_amdgcn_s_wait_asynccnt)
    __builtin_amdgcn_s_wait_asynccnt(0);
  #else
    asm volatile("s_wait_asynccnt 0x0" ::: "memory");
  #endif
#endif
  }
  __syncthreads();

  v8f c0 = {}, c1 = {}, c2 = {}, c3 = {};
  int t0 = wave << 2;                  // first of this wave's 4 column tiles
#pragma unroll 4
  for (int ks = 0; ks < 16; ++ks) {
    v16h a  = load_a_lds(sA, ks << 5);
    v16h b0 = load_b_pk(pkUr, t0 + 0, 16, ks);
    c0 = __builtin_amdgcn_wmma_f32_16x16x32_f16(false, a, false, b0, (short)0, c0, false, false);
    v16h b1 = load_b_pk(pkUr, t0 + 1, 16, ks);
    c1 = __builtin_amdgcn_wmma_f32_16x16x32_f16(false, a, false, b1, (short)0, c1, false, false);
    v16h b2 = load_b_pk(pkUr, t0 + 2, 16, ks);
    c2 = __builtin_amdgcn_wmma_f32_16x16x32_f16(false, a, false, b2, (short)0, c2, false, false);
    v16h b3 = load_b_pk(pkUr, t0 + 3, 16, ks);
    c3 = __builtin_amdgcn_wmma_f32_16x16x32_f16(false, a, false, b3, (short)0, c3, false, false);
  }

  float lacc[8];
#pragma unroll
  for (int r = 0; r < 8; ++r) lacc[r] = 0.f;
  auto acc_tile = [&](const v8f& c, int ct) {
    int col = (wave << 6) + (ct << 4) + (lane & 15);
    float qp = qproj[(size_t)bIdx * D_MODEL + col];
    float ua = Ua[col];
#pragma unroll
    for (int r = 0; r < 8; ++r) lacc[r] += tanhf(c[r] + qp) * ua;
  };
  acc_tile(c0, 0); acc_tile(c1, 1); acc_tile(c2, 2); acc_tile(c3, 3);

  // Reduce over the 16 N-lanes (xor masks 1..8 never cross the 16-lane group).
#pragma unroll
  for (int r = 0; r < 8; ++r) {
    float v = lacc[r];
    v += __shfl_xor(v, 1, 32);
    v += __shfl_xor(v, 2, 32);
    v += __shfl_xor(v, 4, 32);
    v += __shfl_xor(v, 8, 32);
    if ((lane & 15) == 0) atomicAdd(&slog[r + (hlf << 3)], v);
  }
  __syncthreads();
  if (tid < 16) logits[row0 + tid] = slog[tid];
}

// ---------------------------------------------------------------------------
// Kernel 5: masked softmax over N, then out[b,:] = attn @ x_real[b]
// ---------------------------------------------------------------------------
__global__ void softmax_out_kernel(const float* __restrict__ logits, const float* __restrict__ mask,
                                   const _Float16* __restrict__ xreal, float* __restrict__ out) {
  __shared__ float sattn[N_OBJ];
  __shared__ float sred[N_OBJ];
  int b = blockIdx.x, tid = threadIdx.x;                // blockDim = 128
  float l = logits[b * N_OBJ + tid] + (1.0f - mask[b * N_OBJ + tid]) * -1e9f;
  sred[tid] = l;
  __syncthreads();
  for (int s = 64; s > 0; s >>= 1) {
    if (tid < s) sred[tid] = fmaxf(sred[tid], sred[tid + s]);
    __syncthreads();
  }
  float mx = sred[0];
  __syncthreads();
  float e = __expf(l - mx);
  sred[tid] = e;
  __syncthreads();
  for (int s = 64; s > 0; s >>= 1) {
    if (tid < s) sred[tid] += sred[tid + s];
    __syncthreads();
  }
  float inv = 1.0f / sred[0];
  sattn[tid] = e * inv;
  __syncthreads();
  const _Float16* xb = xreal + (size_t)b * N_OBJ * D_MODEL;
#pragma unroll
  for (int g = 0; g < 4; ++g) {
    int col = tid + (g << 7);
    float acc = 0.f;
    for (int n = 0; n < N_OBJ; ++n) acc += sattn[n] * (float)xb[(size_t)n * D_MODEL + col];
    out[(size_t)b * D_MODEL + col] = acc;
  }
}

// ---------------------------------------------------------------------------
extern "C" void kernel_launch(void* const* d_in, const int* in_sizes, int n_in,
                              void* d_out, int out_size, void* d_ws, size_t ws_size,
                              hipStream_t stream) {
  (void)in_sizes; (void)n_in; (void)out_size; (void)ws_size;
  const float* x      = (const float*)d_in[0];
  const float* mask   = (const float*)d_in[1];
  const float* conv_w = (const float*)d_in[2];
  const float* conv_b = (const float*)d_in[3];
  const float* Uq     = (const float*)d_in[4];
  const float* Ur     = (const float*)d_in[5];
  const float* Ua     = (const float*)d_in[6];
  float* out = (float*)d_out;

  char* ws = (char*)d_ws;
  size_t off = 0;
  auto wsalloc = [&](size_t bytes) -> char* {
    char* p = ws + off; off += (bytes + 255) & ~(size_t)255; return p;
  };
  _Float16* pkW    = (_Float16*)wsalloc((size_t)D_OBJ  * D_MODEL * 2);
  _Float16* pkUq   = (_Float16*)wsalloc((size_t)D_MODEL * D_MODEL * 2);
  _Float16* pkUr   = (_Float16*)wsalloc((size_t)D_MODEL * D_MODEL * 2);
  _Float16* xreal  = (_Float16*)wsalloc((size_t)B_DIM * N_OBJ * D_MODEL * 2);
  float*    query  = (float*)wsalloc((size_t)B_DIM * D_MODEL * 4);
  float*    qproj  = (float*)wsalloc((size_t)B_DIM * D_MODEL * 4);
  float*    logits = (float*)wsalloc((size_t)B_DIM * N_OBJ * 4);

  // Pre-pack weights into WMMA B-fragment layout (f32 -> f16).
  pack_b_kernel<<<dim3((2048  + 255) / 256), 256, 0, stream>>>(conv_w, pkW, D_OBJ,   D_MODEL);
  pack_b_kernel<<<dim3((16384 + 255) / 256), 256, 0, stream>>>(Uq,     pkUq, D_MODEL, D_MODEL);
  pack_b_kernel<<<dim3((16384 + 255) / 256), 256, 0, stream>>>(Ur,     pkUr, D_MODEL, D_MODEL);

  emb_mask_kernel<<<dim3(B_DIM * N_OBJ / 16, D_MODEL / 128), 256, 0, stream>>>(x, mask, conv_b, pkW, xreal);
  query_kernel<<<dim3(B_DIM), 256, 0, stream>>>(xreal, mask, query);
  qproj_kernel<<<dim3(B_DIM / 16, D_MODEL / 128), 256, 0, stream>>>(query, pkUq, qproj);
  logits_kernel<<<dim3(B_DIM * N_OBJ / 16), 256, 0, stream>>>(xreal, pkUr, qproj, Ua, logits);
  softmax_out_kernel<<<dim3(B_DIM), 128, 0, stream>>>(logits, mask, xreal, out);
}
